// FlashPhiAttention_58660663329443
// MI455X (gfx1250) — matrix operand order repaired and checked by reading, not compiled, not run on hardware
//
#include <hip/hip_runtime.h>
#include <hip/hip_bf16.h>

// ---------------------------------------------------------------------------
// FlashPhiAttention for MI455X (gfx1250, wave32, WMMA, bf16 datapath)
//   hidden -> [cvt bf16] -> QKV GEMM (q/k fp32, v transposed bf16)
//   -> fused RoPE+cvt (qk bf16) -> causal flash attn (bf16 out)
//   -> dense GEMM -> out fp32
// GEMMs: A via global_load_b128, B tile staged through LDS by the Tensor
// Data Mover (TENSOR_LOAD_TO_LDS, double buffered, s_wait_tensorcnt), math
// on v_wmma_f32_16x16x32_bf16. No fp32->bf16 VALU in the inner loops.
// ---------------------------------------------------------------------------

#define T_TOK   8192
#define SEQ     1024
#define BATCH   8
#define HID     2560
#define NHEAD   32
#define DHEAD   80
#define QKV_N   7680          // (H + 2*KV) * D
#define QK_N    5120          // q + k columns
#define ROT_H   20            // ROT/2
#define ATT_SCALE 0.08838834764831845f  // 1/sqrt(128)

typedef __attribute__((ext_vector_type(16))) __bf16 v16bf;
typedef __attribute__((ext_vector_type(8)))  __bf16 bf16x8;
typedef __attribute__((ext_vector_type(8)))  float  v8f;

typedef unsigned int tdm_u32x4 __attribute__((ext_vector_type(4)));
typedef int          tdm_i32x4 __attribute__((ext_vector_type(4)));
typedef int          tdm_i32x8 __attribute__((ext_vector_type(8)));

// Device pass: builtins exist -> real TDM path. Host pass: stubs (host never
// codegens device bodies, but it must parse them).
#if __has_builtin(__builtin_amdgcn_tensor_load_to_lds) && \
    __has_builtin(__builtin_amdgcn_s_wait_tensorcnt)
#define USE_TDM 1
#else
#define USE_TDM 0
#endif

__device__ __forceinline__ unsigned short f2bf_bits(float f) {
  unsigned u = __builtin_bit_cast(unsigned, f);
  u += 0x7FFFu + ((u >> 16) & 1u);          // round-to-nearest-even
  return (unsigned short)(u >> 16);
}
__device__ __forceinline__ __bf16 bits2bf(unsigned short h) {
  return __builtin_bit_cast(__bf16, h);
}
__device__ __forceinline__ __bf16 f2bf(float f) { return bits2bf(f2bf_bits(f)); }

__device__ __forceinline__ v16bf mk16(bf16x8 lo, bf16x8 hi) {
  return __builtin_shufflevector(lo, hi, 0,1,2,3,4,5,6,7,8,9,10,11,12,13,14,15);
}

// A-matrix (16x32 bf16) per-lane element->K map (contiguous, matches B):
// lanes 0-15:  elems 0..15 -> K 0..15
// lanes 16-31: elems 0..15 -> K 16..31
__device__ __forceinline__ int a_kmap(int i, int half) {
  return i + half * 16;
}

#if USE_TDM
// Issue one TDM 2D tile load: tile_d1 rows x tile_d0 bf16 elems, row stride
// stride_d0 (elems) in global, packed contiguously into LDS at lds_off.
// D# layout per CDNA5 ISA section 8 (groups 2/3 zero: 2D tensor).
__device__ __forceinline__ void tdm_load_2d(unsigned lds_off, const void* gaddr,
                                            unsigned tile_d0, unsigned tile_d1,
                                            unsigned stride_d0) {
  const unsigned long long ga = (unsigned long long)gaddr;
  tdm_u32x4 g0;
  g0[0] = 1u;                                     // count=1, user descriptor
  g0[1] = lds_off;                                // lds_addr (bytes)
  g0[2] = (unsigned)(ga & 0xFFFFFFFFu);           // global_addr[31:0]
  g0[3] = (unsigned)((ga >> 32) & 0x01FFFFFFu)    // global_addr[56:32]
          | (2u << 30);                           // type=2 ("image")
  tdm_i32x8 g1;
  g1[0] = (int)(1u << 16);                        // data_size=1 (2B), mask=0
  g1[1] = (int)((tile_d0 & 0xFFFFu) << 16);       // tensor_dim0[15:0]
  g1[2] = (int)(((tile_d0 >> 16) & 0xFFFFu)       // tensor_dim0[31:16]
          | ((tile_d1 & 0xFFFFu) << 16));         // tensor_dim1[15:0]
  g1[3] = (int)(((tile_d1 >> 16) & 0xFFFFu)       // tensor_dim1[31:16]
          | ((tile_d0 & 0xFFFFu) << 16));         // tile_dim0
  g1[4] = (int)(tile_d1 & 0xFFFFu);               // tile_dim1 (tile_dim2=0)
  g1[5] = (int)stride_d0;                         // tensor_dim0_stride[31:0]
  g1[6] = 0;                                      // stride0[47:32], stride1 lo
  g1[7] = 0;                                      // stride1 hi
  tdm_i32x4 gz = {0, 0, 0, 0};
#if defined(__clang_major__) && (__clang_major__ >= 23)
  tdm_i32x8 gz8 = {0, 0, 0, 0, 0, 0, 0, 0};
  __builtin_amdgcn_tensor_load_to_lds(g0, g1, gz, gz, gz8, 0);
#else
  __builtin_amdgcn_tensor_load_to_lds(g0, g1, gz, gz, 0);
#endif
}
__device__ __forceinline__ void tdm_wait() {
  __builtin_amdgcn_s_wait_tensorcnt(0);
}
#else
// Host-pass / no-TDM stubs (parse-only; device fallback uses sync copies).
__device__ __forceinline__ void tdm_load_2d(unsigned, const void*, unsigned,
                                            unsigned, unsigned) {}
__device__ __forceinline__ void tdm_wait() {}
#endif

// ---------------------------------------------------------------------------
// Elementwise fp32 -> bf16 (weights / activations pre-pass). n % 8 == 0.
// ---------------------------------------------------------------------------
__global__ __launch_bounds__(256)
void cvt_f32_bf16(const float* __restrict__ src, __bf16* __restrict__ dst,
                  long n) {
  const long i = ((long)blockIdx.x * blockDim.x + threadIdx.x) * 8;
  if (i >= n) return;
  const float4 a = *(const float4*)(src + i);
  const float4 b = *(const float4*)(src + i + 4);
  bf16x8 o;
  o[0] = f2bf(a.x); o[1] = f2bf(a.y); o[2] = f2bf(a.z); o[3] = f2bf(a.w);
  o[4] = f2bf(b.x); o[5] = f2bf(b.y); o[6] = f2bf(b.z); o[7] = f2bf(b.w);
  *(bf16x8*)(dst + i) = o;
}

// ---------------------------------------------------------------------------
// GEMM core: C = A[M,K](bf16) * W[N,K](bf16)^T, f32 accumulate.
// 256 threads = 8 waves; block tile 128x128; wave(wm 0..3, wn 0..1) owns
// 32x64 = 2x4 tiles. A: global_load_b128. B: 128x32 bf16 tile (8 KB),
// double-buffered in LDS, filled by TDM (wave 0), consumed via ds_load.
// ---------------------------------------------------------------------------
#define GEMM_FILL_B(W_, K_, bufidx, kkn)                                       \
  do {                                                                         \
    if (USE_TDM_LOCAL) {                                                       \
      if (wave == 0)                                                           \
        tdm_load_2d((unsigned)(unsigned long long)&ldsB[bufidx][0],            \
                    W_ + (size_t)bnBlk * K_ + (kkn), 32u, 128u, (unsigned)K_); \
    } else {                                                                   \
      const int nl = tid >> 1, seg = tid & 1;                                  \
      const __bf16* gs = W_ + (size_t)(bnBlk + nl) * K_ + (kkn) + seg * 16;    \
      *(bf16x8*)&ldsB[bufidx][nl * 32 + seg * 16] = *(const bf16x8*)gs;        \
      *(bf16x8*)&ldsB[bufidx][nl * 32 + seg * 16 + 8] =                        \
          *(const bf16x8*)(gs + 8);                                            \
    }                                                                          \
  } while (0)

#define GEMM_CORE(A_, W_, K_)                                                  \
  __shared__ __bf16 ldsB[2][128 * 32]; /* 16 KB double buffer */               \
  const int tid = threadIdx.x;                                                 \
  const int lane = tid & 31;                                                   \
  const int wave = tid >> 5;                                                   \
  const int wm = wave >> 1;                                                    \
  const int wn = wave & 1;                                                     \
  const int half = lane >> 4;                                                  \
  const int l16 = lane & 15;                                                   \
  const int bm = blockIdx.y * 128 + wm * 32;                                   \
  const int bnBlk = blockIdx.x * 128;                                          \
  const int bn = bnBlk + wn * 64;                                              \
  const int USE_TDM_LOCAL = USE_TDM;                                           \
  v8f vzero = {};                                                              \
  v8f acc[2][4];                                                               \
  _Pragma("unroll") for (int mt = 0; mt < 2; ++mt)                             \
      _Pragma("unroll") for (int nt = 0; nt < 4; ++nt) acc[mt][nt] = vzero;    \
  const size_t aRow0 = (size_t)(bm + l16) * K_ + half * 8;                     \
  const size_t aRow1 = (size_t)(bm + 16 + l16) * K_ + half * 8;                \
  GEMM_FILL_B(W_, K_, 0, 0);                                                   \
  int cur = 0;                                                                 \
  for (int kk = 0; kk < K_; kk += 32) {                                        \
    if (USE_TDM_LOCAL && wave == 0) tdm_wait();                                \
    __syncthreads(); /* buf[cur] visible to all waves */                       \
    if (kk + 32 < K_) {                                                        \
      GEMM_FILL_B(W_, K_, cur ^ 1, kk + 32);                                   \
      __builtin_prefetch(A_ + aRow0 + kk + 32, 0, 1);                          \
    }                                                                          \
    v16bf af[2];                                                               \
    af[0] = mk16(*(const bf16x8*)(A_ + aRow0 + kk),                            \
                 *(const bf16x8*)(A_ + aRow0 + kk + 16));                      \
    af[1] = mk16(*(const bf16x8*)(A_ + aRow1 + kk),                            \
                 *(const bf16x8*)(A_ + aRow1 + kk + 16));                      \
    v16bf bfr[4];                                                              \
    _Pragma("unroll") for (int nt = 0; nt < 4; ++nt)                           \
        bfr[nt] = mk16(*(const bf16x8*)&ldsB[cur][(wn * 64 + nt * 16 + l16) * 32 + half * 8], \
                       *(const bf16x8*)&ldsB[cur][(wn * 64 + nt * 16 + l16) * 32 + 16 + half * 8]); \
    _Pragma("unroll") for (int mt = 0; mt < 2; ++mt)                           \
        _Pragma("unroll") for (int nt = 0; nt < 4; ++nt)                       \
            acc[mt][nt] = __builtin_amdgcn_wmma_f32_16x16x32_bf16(             \
                false, af[mt], false, bfr[nt], (short)0, acc[mt][nt], false,   \
                false);                                                        \
    cur ^= 1;                                                                  \
  }

// Standard GEMM: fp32 output + bias (used for dense projection).
__global__ __launch_bounds__(256)
void wmma_gemm_bias(const __bf16* __restrict__ A, const __bf16* __restrict__ W,
                    const float* __restrict__ bias, float* __restrict__ C,
                    int M, int N, int K) {
  GEMM_CORE(A, W, K)
#pragma unroll
  for (int nt = 0; nt < 4; ++nt) {
    const int col = bn + nt * 16 + l16;
    const float bv = bias[col];
#pragma unroll
    for (int mt = 0; mt < 2; ++mt)
#pragma unroll
      for (int j = 0; j < 8; ++j) {
        const int row = bm + mt * 16 + j + half * 8;
        C[(size_t)row * N + col] = acc[mt][nt][j] + bv;
      }
  }
}

// QKV GEMM: q/k columns -> fp32 buffer [T, 5120] (RoPE needs fp32 RMW);
// v columns -> bf16, stored transposed as vT[b][h][d][s] for PV fragments.
__global__ __launch_bounds__(256)
void wmma_gemm_qkv(const __bf16* __restrict__ A, const __bf16* __restrict__ W,
                   const float* __restrict__ bias, float* __restrict__ qk,
                   __bf16* __restrict__ vT, int M, int N, int K) {
  GEMM_CORE(A, W, K)
#pragma unroll
  for (int nt = 0; nt < 4; ++nt) {
    const int col = bn + nt * 16 + l16;
    const float bv = bias[col];
    if (col < QK_N) {                 // uniform per 16-wide tile (5120 % 16 == 0)
#pragma unroll
      for (int mt = 0; mt < 2; ++mt)
#pragma unroll
        for (int j = 0; j < 8; ++j) {
          const int row = bm + mt * 16 + j + half * 8;
          qk[(size_t)row * QK_N + col] = acc[mt][nt][j] + bv;
        }
    } else {
      const int hd = col - QK_N;
      const int h = hd / DHEAD;
      const int d = hd % DHEAD;
#pragma unroll
      for (int mt = 0; mt < 2; ++mt)
#pragma unroll
        for (int j = 0; j < 8; ++j) {
          const int row = bm + mt * 16 + j + half * 8;
          const int bb = row >> 10;          // row / SEQ
          const int ss = row & (SEQ - 1);    // row % SEQ
          vT[((size_t)(bb * NHEAD + h) * DHEAD + d) * SEQ + ss] =
              f2bf(acc[mt][nt][j] + bv);
        }
    }
  }
}

// ---------------------------------------------------------------------------
// Fused partial RoPE + fp32->bf16: qk_f32[T,5120] -> qk_bf16[T,5120]
// ---------------------------------------------------------------------------
__global__ __launch_bounds__(256)
void rope_cvt(const float* __restrict__ qk, const float* __restrict__ cosp,
              const float* __restrict__ sinp, __bf16* __restrict__ out) {
  const long idx = (long)blockIdx.x * blockDim.x + threadIdx.x;
  if (idx >= (long)T_TOK * QK_N) return;
  const int t = (int)(idx / QK_N);
  const int j = (int)(idx % QK_N);
  const int head = j / DHEAD;          // 0..63 (q heads then k heads)
  const int din = j % DHEAD;
  const float* row = qk + (size_t)t * QK_N + head * DHEAD;
  float v;
  if (din < ROT_H) {
    const float c = cosp[(size_t)t * ROT_H + din];
    const float s = sinp[(size_t)t * ROT_H + din];
    v = row[din] * c - row[din + ROT_H] * s;
  } else if (din < 2 * ROT_H) {
    const int i = din - ROT_H;
    const float c = cosp[(size_t)t * ROT_H + i];
    const float s = sinp[(size_t)t * ROT_H + i];
    v = row[din] * c + row[i] * s;
  } else {
    v = row[din];
  }
  out[idx] = f2bf(v);
}

// ---------------------------------------------------------------------------
// Causal flash attention: one wave per (b, h, 16-row q block).
// Q/K from qk_bf16[T,5120]; V from vT[b][h][d][s]; 32 keys per inner block.
// QK^T: 3 d-chunks of 32 (80 padded to 96). PV: 5 d-chunks of 16.
// ---------------------------------------------------------------------------
__global__ __launch_bounds__(32)
void flash_attn(const __bf16* __restrict__ qk, const __bf16* __restrict__ vT,
                __bf16* __restrict__ attn) {
  const int lane = threadIdx.x & 31;
  const int half = lane >> 4;
  const int l16  = lane & 15;
  const int qb = blockIdx.x;   // 0..63
  const int h  = blockIdx.y;   // 0..31
  const int b  = blockIdx.z;   // 0..7

  __shared__ unsigned short pbuf[16 * 32];  // P transpose staging (bf16 bits)

  const v16bf v16z = {};
  const bf16x8 v8z = {};

  // ---- Q fragments (A layout): b128 pairs, chunk 2 hi-half zero padded ----
  v16bf qf[3];
  {
    const int trow = b * SEQ + qb * 16 + l16;
    const __bf16* qp = qk + (size_t)trow * QK_N + h * DHEAD;
#pragma unroll
    for (int ci = 0; ci < 3; ++ci) {
      const bf16x8 lo = *(const bf16x8*)(qp + ci * 32 + half * 8);
      const bf16x8 hi = (ci < 2) ? *(const bf16x8*)(qp + ci * 32 + 16 + half * 8)
                                 : v8z;
      qf[ci] = mk16(lo, hi);
    }
  }

  v8f vzero = {};
  v8f oacc[5];
#pragma unroll
  for (int dc = 0; dc < 5; ++dc) oacc[dc] = vzero;
  float mrun[8], lrun[8];
#pragma unroll
  for (int j = 0; j < 8; ++j) { mrun[j] = -3.0e38f; lrun[j] = 0.0f; }

  const int qlast = qb * 16 + 15;
  const int nkb = qlast / 32 + 1;

  for (int kb = 0; kb < nkb; ++kb) {
    const int k0 = kb * 32;

    // ---- scores S[16q x 32k] as two 16x16 tiles ----
    v8f sc[2];
#pragma unroll
    for (int tile = 0; tile < 2; ++tile) {
      v8f s = vzero;
      const int key = k0 + tile * 16 + l16;        // B column for this lane
      const __bf16* kp = qk + (size_t)(b * SEQ + key) * QK_N
                            + NHEAD * DHEAD + h * DHEAD;
#pragma unroll
      for (int ci = 0; ci < 3; ++ci) {
        // B-fragment K map: same table order as Q (half*8 / 16+half*8)
        v16bf kf = mk16(*(const bf16x8*)(kp + ci * 32 + half * 8),
                        (ci < 2) ? *(const bf16x8*)(kp + ci * 32 + 16 + half * 8)
                                 : v8z);           // d >= 80 padding
        s = __builtin_amdgcn_wmma_f32_16x16x32_bf16(
            false, qf[ci], false, kf, (short)0, s, false, false);
      }
      sc[tile] = s;
    }

    // ---- scale + causal mask + online softmax update ----
    float pf0[8], pf1[8], alpha[8];
#pragma unroll
    for (int j = 0; j < 8; ++j) {
      const int qrow = qb * 16 + j + half * 8;
      float s0 = sc[0][j] * ATT_SCALE;
      float s1 = sc[1][j] * ATT_SCALE;
      if (k0 + l16 > qrow)      s0 = -3.0e38f;
      if (k0 + 16 + l16 > qrow) s1 = -3.0e38f;
      float mx = fmaxf(s0, s1);
      mx = fmaxf(mx, __shfl_xor(mx, 1, 32));
      mx = fmaxf(mx, __shfl_xor(mx, 2, 32));
      mx = fmaxf(mx, __shfl_xor(mx, 4, 32));
      mx = fmaxf(mx, __shfl_xor(mx, 8, 32));
      const float mn = fmaxf(mrun[j], mx);
      const float p0 = __expf(s0 - mn);
      const float p1 = __expf(s1 - mn);
      float rs = p0 + p1;
      rs += __shfl_xor(rs, 1, 32);
      rs += __shfl_xor(rs, 2, 32);
      rs += __shfl_xor(rs, 4, 32);
      rs += __shfl_xor(rs, 8, 32);
      alpha[j] = __expf(mrun[j] - mn);
      mrun[j] = mn;
      lrun[j] = lrun[j] * alpha[j] + rs;
      pf0[j] = p0; pf1[j] = p1;
    }

    // ---- transpose P through LDS into A-fragment layout ----
#pragma unroll
    for (int j = 0; j < 8; ++j) {
      const int row = j + half * 8;
      pbuf[row * 32 + l16]      = f2bf_bits(pf0[j]);
      pbuf[row * 32 + 16 + l16] = f2bf_bits(pf1[j]);
    }
    __syncthreads();
    v16bf pA;
#pragma unroll
    for (int i = 0; i < 16; ++i)
      pA[i] = bits2bf(pbuf[l16 * 32 + a_kmap(i, half)]);
    __syncthreads();

    // ---- rescale O, then O += P * V (5 d-chunks of 16) ----
#pragma unroll
    for (int dc = 0; dc < 5; ++dc)
#pragma unroll
      for (int j = 0; j < 8; ++j) oacc[dc][j] *= alpha[j];

#pragma unroll
    for (int dc = 0; dc < 5; ++dc) {
      const int d = dc * 16 + l16;                 // B column = head dim
      // B-fragment K map: kv = half*16 + i -> contiguous along s in vT
      const v16bf vf = *(const v16bf*)(
          vT + ((size_t)(b * NHEAD + h) * DHEAD + d) * SEQ + k0 + half * 16);
      oacc[dc] = __builtin_amdgcn_wmma_f32_16x16x32_bf16(
          false, pA, false, vf, (short)0, oacc[dc], false, false);
    }
  }

  // ---- normalize + write attn[T, HID] as bf16 (dense GEMM A operand) ----
#pragma unroll
  for (int dc = 0; dc < 5; ++dc) {
    const int d = dc * 16 + l16;
#pragma unroll
    for (int j = 0; j < 8; ++j) {
      const int t = b * SEQ + qb * 16 + j + half * 8;
      attn[(size_t)t * HID + h * DHEAD + d] = f2bf(oacc[dc][j] / lrun[j]);
    }
  }
}

// ---------------------------------------------------------------------------
extern "C" void kernel_launch(void* const* d_in, const int* in_sizes, int n_in,
                              void* d_out, int out_size, void* d_ws, size_t ws_size,
                              hipStream_t stream) {
  const float* hs      = (const float*)d_in[0];
  const float* cosp    = (const float*)d_in[1];
  const float* sinp    = (const float*)d_in[2];
  const float* w_qkv   = (const float*)d_in[3];
  const float* b_qkv   = (const float*)d_in[4];
  const float* w_dense = (const float*)d_in[5];
  const float* b_dense = (const float*)d_in[6];
  float* out = (float*)d_out;

  // Workspace layout (bf16 elements unless noted):
  __bf16* hs_bf   = (__bf16*)d_ws;                      // [8192,2560]
  __bf16* wqkv_bf = hs_bf   + (size_t)T_TOK * HID;      // [7680,2560]
  __bf16* wd_bf   = wqkv_bf + (size_t)QKV_N * HID;      // [2560,2560]
  __bf16* qk_bf   = wd_bf   + (size_t)HID * HID;        // [8192,5120]
  __bf16* vT      = qk_bf   + (size_t)T_TOK * QK_N;     // [8,32,80,1024]
  float*  qk_f32  = (float*)(vT + (size_t)T_TOK * HID); // [8192,5120] fp32
  __bf16* attn_bf = hs_bf;  // alias: hs_bf dead after QKV GEMM

  // 0) one-shot fp32 -> bf16 conversions (memory-bound pre-pass)
  {
    const long n_hs = (long)T_TOK * HID;
    const long n_wq = (long)QKV_N * HID;
    const long n_wd = (long)HID * HID;
    cvt_f32_bf16<<<(unsigned)((n_hs / 8 + 255) / 256), 256, 0, stream>>>(
        hs, hs_bf, n_hs);
    cvt_f32_bf16<<<(unsigned)((n_wq / 8 + 255) / 256), 256, 0, stream>>>(
        w_qkv, wqkv_bf, n_wq);
    cvt_f32_bf16<<<(unsigned)((n_wd / 8 + 255) / 256), 256, 0, stream>>>(
        w_dense, wd_bf, n_wd);
  }

  // 1) QKV GEMM: q/k -> fp32, v -> bf16 transposed
  wmma_gemm_qkv<<<dim3(QKV_N / 128, T_TOK / 128), 256, 0, stream>>>(
      hs_bf, wqkv_bf, b_qkv, qk_f32, vT, T_TOK, QKV_N, HID);

  // 2) fused partial RoPE + bf16 convert for q/k
  {
    const long total = (long)T_TOK * QK_N;
    rope_cvt<<<(unsigned)((total + 255) / 256), 256, 0, stream>>>(
        qk_f32, cosp, sinp, qk_bf);
  }

  // 3) causal flash attention (writes bf16 attn)
  flash_attn<<<dim3(SEQ / 16, NHEAD, BATCH), 32, 0, stream>>>(
      qk_bf, vT, attn_bf);

  // 4) out = attn @ w_dense^T + b_dense (fp32 out)
  wmma_gemm_bias<<<dim3(HID / 128, T_TOK / 128), 256, 0, stream>>>(
      attn_bf, wd_bf, b_dense, out, T_TOK, HID, HID);
}
